// SparseLinearAttention_9938554323658
// MI455X (gfx1250) — matrix operand order repaired and checked by reading, not compiled, not run on hardware
//
#include <hip/hip_runtime.h>
#include <hip/hip_bf16.h>

// Problem constants (from reference): B=2, N=2048, C=1024, H=16, D=64, topk=102
#define BB 2
#define NN 2048
#define CC 1024
#define HH 16
#define DD 64
#define TOPK 102

typedef __bf16 bf16_t;
typedef __attribute__((ext_vector_type(16))) __bf16 v16bf;
typedef __attribute__((ext_vector_type(8)))  float   v8f;
typedef __attribute__((ext_vector_type(4)))  unsigned u32x4;
typedef __attribute__((ext_vector_type(8)))  int      i32x8;
typedef __attribute__((ext_vector_type(4)))  int      i32x4;
typedef unsigned long long u64;

// ---------------- wave32 reduction helpers ----------------
__device__ __forceinline__ float wave_sum_f(float v) {
#pragma unroll
  for (int m = 16; m >= 1; m >>= 1) v += __shfl_xor(v, m, 32);
  return v;
}
__device__ __forceinline__ float wave_max_f(float v) {
#pragma unroll
  for (int m = 16; m >= 1; m >>= 1) v = fmaxf(v, __shfl_xor(v, m, 32));
  return v;
}
__device__ __forceinline__ float wave_min_f(float v) {
#pragma unroll
  for (int m = 16; m >= 1; m >>= 1) v = fminf(v, __shfl_xor(v, m, 32));
  return v;
}
__device__ __forceinline__ int wave_sum_i(int v) {
#pragma unroll
  for (int m = 16; m >= 1; m >>= 1) v += __shfl_xor(v, m, 32);
  return v;
}

// order-preserving float <-> uint mapping (for exact k-th largest search)
__device__ __forceinline__ unsigned f2ord(float f) {
  unsigned u = __float_as_uint(f);
  return (u & 0x80000000u) ? ~u : (u | 0x80000000u);
}
__device__ __forceinline__ float ord2f(unsigned o) {
  unsigned u = (o & 0x80000000u) ? (o & 0x7FFFFFFFu) : ~o;
  return __uint_as_float(u);
}

// ---------------- CDNA5 async global->LDS copy (ASYNCcnt path) ----------
// LDS destination is the wave-relative LDS byte address (= low 32 bits of the
// generic pointer to a __shared__ object on gfx1250).
__device__ __forceinline__ void async_copy_b128(unsigned lds_off, u64 gaddr) {
  asm volatile("global_load_async_to_lds_b128 %0, %1, off"
               :: "v"(lds_off), "v"(gaddr) : "memory");
}
__device__ __forceinline__ void wait_async_le4() {
#if __has_builtin(__builtin_amdgcn_s_wait_asynccnt)
  __builtin_amdgcn_s_wait_asynccnt(4);
#else
  asm volatile("s_wait_asynccnt 0x4" ::: "memory");
#endif
}
__device__ __forceinline__ void wait_async_le0() {
#if __has_builtin(__builtin_amdgcn_s_wait_asynccnt)
  __builtin_amdgcn_s_wait_asynccnt(0);
#else
  asm volatile("s_wait_asynccnt 0x0" ::: "memory");
#endif
}

// ---------------- elementwise utility kernels ----------------
__global__ void cvt_f32_to_bf16_kernel(const float* __restrict__ in,
                                       bf16_t* __restrict__ out, int n) {
  for (int i = blockIdx.x * 256 + threadIdx.x; i < n; i += gridDim.x * 256)
    out[i] = (bf16_t)in[i];
}

__global__ void zero_f32_kernel(float* __restrict__ p, int n) {
  int i = blockIdx.x * 256 + threadIdx.x;
  if (i < n) p[i] = 0.0f;
}

// ---------------- K1: QKV GEMM (bf16 WMMA, async double-buffered LDS) ----
// X (4096 x 1024) @ Wqkv^T (1024 x 3072) + bqkv, scattered into (B,H,N,D)
// layouts for q,k,v (fp32) and v transposed per head as bf16 (D x N).
__global__ __launch_bounds__(256)
void qkv_gemm_kernel(const bf16_t* __restrict__ Xb, const bf16_t* __restrict__ Wb,
                     const float* __restrict__ bias,
                     float* __restrict__ qbuf, float* __restrict__ kbuf,
                     float* __restrict__ vbuf, bf16_t* __restrict__ vTb) {
  __shared__ __align__(16) bf16_t As[2][128 * 32];
  __shared__ __align__(16) bf16_t Bs[2][128 * 32];   // stored [col][k]
  const int tid  = threadIdx.x;
  const int wave = tid >> 5, lane = tid & 31;
  const int half = lane >> 4, l16 = lane & 15;
  const int rowBase = blockIdx.x * 128;
  const int colBase = blockIdx.y * 128;
  v8f acc[8] = {};

  const int cr = tid >> 1;          // row (A) / col (B) copied by this thread
  const int ch = (tid & 1) * 16;    // 16-element half of the 32-wide K slab
  const bf16_t* gA = Xb + (size_t)(rowBase + cr) * CC + ch;
  const bf16_t* gB = Wb + (size_t)(colBase + cr) * CC + ch;
  const unsigned ldsA0 = (unsigned)(uintptr_t)(&As[0][cr * 32 + ch]);
  const unsigned ldsB0 = (unsigned)(uintptr_t)(&Bs[0][cr * 32 + ch]);
  const unsigned BUFB = 128u * 32u * 2u;   // bytes per LDS buffer

  // prologue: stage slab 0 asynchronously (2 x b128 per tile per thread)
  async_copy_b128(ldsA0,      (u64)(uintptr_t)gA);
  async_copy_b128(ldsA0 + 16, (u64)(uintptr_t)(gA + 8));
  async_copy_b128(ldsB0,      (u64)(uintptr_t)gB);
  async_copy_b128(ldsB0 + 16, (u64)(uintptr_t)(gB + 8));

  for (int it = 0; it < 32; ++it) {
    const int buf = it & 1;
    if (it + 1 < 32) {   // stream next K-slab while computing this one
      const bf16_t* nA = gA + (size_t)(it + 1) * 32;
      const bf16_t* nB = gB + (size_t)(it + 1) * 32;
      const unsigned la = ldsA0 + (unsigned)(buf ^ 1) * BUFB;
      const unsigned lb = ldsB0 + (unsigned)(buf ^ 1) * BUFB;
      async_copy_b128(la,      (u64)(uintptr_t)nA);
      async_copy_b128(la + 16, (u64)(uintptr_t)(nA + 8));
      async_copy_b128(lb,      (u64)(uintptr_t)nB);
      async_copy_b128(lb + 16, (u64)(uintptr_t)(nB + 8));
      wait_async_le4();   // oldest 4 (current slab) complete in order
    } else {
      wait_async_le0();
    }
    __syncthreads();

    v16bf af;
    const bf16_t* ap = &As[buf][(16 * wave + l16) * 32];
#pragma unroll
    for (int j = 0; j < 16; ++j) af[j] = ap[16 * half + j];
#pragma unroll
    for (int ct = 0; ct < 8; ++ct) {
      v16bf bfm;
      const bf16_t* bp = &Bs[buf][(16 * ct + l16) * 32 + 16 * half];
#pragma unroll
      for (int j = 0; j < 16; ++j) bfm[j] = bp[j];
      acc[ct] = __builtin_amdgcn_wmma_f32_16x16x32_bf16(
          false, af, false, bfm, (short)0, acc[ct], false, false);
    }
    __syncthreads();   // protect buffer refilled next iteration
  }

  // epilogue: bias + scatter to q/k/v (B,H,N,D) and vT (B,H,D,N) bf16
#pragma unroll
  for (int ct = 0; ct < 8; ++ct) {
#pragma unroll
    for (int r = 0; r < 8; ++r) {
      int row_g = rowBase + 16 * wave + r + 8 * half;      // 0..4095 = b*N+n
      int col_g = colBase + 16 * ct + l16;                 // 0..3071
      float val = acc[ct][r] + bias[col_g];
      int which = col_g >> 10;                             // 0=q,1=k,2=v
      int cc = col_g & 1023;
      int h = cc >> 6, d = cc & 63;
      int b = row_g >> 11, n = row_g & 2047;
      size_t bh = (size_t)(b * HH + h);
      size_t idx = (bh * NN + n) * DD + d;
      if (which == 0)      qbuf[idx] = val;
      else if (which == 1) kbuf[idx] = val;
      else {
        vbuf[idx] = val;
        vTb[(bh * DD + d) * NN + n] = (bf16_t)val;         // per-head transpose
      }
    }
  }
}

// ---------------- K2: LayerNorm + softmax feature map (one row / wave) ----
__global__ __launch_bounds__(256)
void ln_feat_kernel(const float* __restrict__ in, const float* __restrict__ gamma,
                    const float* __restrict__ beta, bf16_t* __restrict__ out_bf,
                    float* __restrict__ feat, float* __restrict__ kfsum, int doSum) {
  const int wave = threadIdx.x >> 5, lane = threadIdx.x & 31;
  const size_t row = (size_t)blockIdx.x * 8 + wave;        // < B*H*N
  const float* src = in + row * DD;
  float x0 = src[lane], x1 = src[lane + 32];
  float mu = wave_sum_f(x0 + x1) * (1.0f / 64.0f);
  float d0 = x0 - mu, d1 = x1 - mu;
  float var = wave_sum_f(d0 * d0 + d1 * d1) * (1.0f / 64.0f);   // biased
  float inv = rsqrtf(var + 1e-5f);
  float y0 = d0 * inv * gamma[lane] + beta[lane];
  float y1 = d1 * inv * gamma[lane + 32] + beta[lane + 32];
  out_bf[row * DD + lane]      = (bf16_t)y0;
  out_bf[row * DD + lane + 32] = (bf16_t)y1;
  // softmax feature map over D
  float mx = wave_max_f(fmaxf(y0, y1));
  float e0 = __expf(y0 - mx), e1 = __expf(y1 - mx);
  float s = wave_sum_f(e0 + e1);
  float f0 = e0 / s, f1 = e1 / s;
  feat[row * DD + lane]      = f0;
  feat[row * DD + lane + 32] = f1;
  if (doSum) {   // accumulate kf.sum(axis=n) per (b,h,d)
    int bh = (int)(row >> 11);
    atomicAdd(&kfsum[bh * DD + lane], f0);
    atomicAdd(&kfsum[bh * DD + lane + 32], f1);
  }
}

// ---------------- K3: kv = kf^T @ v per head (small D x D GEMM) ---------
__global__ __launch_bounds__(256)
void kv_kernel(const float* __restrict__ kf, const float* __restrict__ v,
               float* __restrict__ kv) {
  __shared__ float kfc[64][64];
  __shared__ float vc[64][64];
  const int bh = blockIdx.x;
  const int t = threadIdx.x;
  const int d0 = t >> 6, e = t & 63;
  float acc[16] = {};
  const float* kfb = kf + (size_t)bh * NN * DD;
  const float* vb  = v  + (size_t)bh * NN * DD;
  for (int n0 = 0; n0 < NN; n0 += 64) {
    for (int idx = t; idx < 64 * 64; idx += 256) {
      int r = idx >> 6, c = idx & 63;
      kfc[r][c] = kfb[(size_t)(n0 + r) * DD + c];
      vc[r][c]  = vb [(size_t)(n0 + r) * DD + c];
    }
    __syncthreads();
#pragma unroll 4
    for (int nn = 0; nn < 64; ++nn) {
      float vv = vc[nn][e];
#pragma unroll
      for (int i = 0; i < 16; ++i) acc[i] += kfc[nn][d0 + 4 * i] * vv;
    }
    __syncthreads();
  }
#pragma unroll
  for (int i = 0; i < 16; ++i)
    kv[((size_t)bh * DD + d0 + 4 * i) * DD + e] = acc[i];
}

// ---------------- K4: fused scores -> softmax -> topk -> PV -> mix -------
// One workgroup (4 waves) per 16-row tile per head. The 16x2048 score strip
// lives entirely in CDNA5's 320KB WGP LDS (137KB used) -> the N x N matrix
// never touches HBM. Q tile is staged by the Tensor Data Mover.
#define SPARSE_SMEM (16 * 2048 * 4 + 16 * 64 * 2 + 16 * 64 * 4)
__global__ __launch_bounds__(128)
void sparse_attn_kernel(const bf16_t* __restrict__ q_bf, const bf16_t* __restrict__ k_bf,
                        const bf16_t* __restrict__ vTb, const float* __restrict__ qf,
                        const float* __restrict__ kv, const float* __restrict__ kfsum,
                        const float* __restrict__ alpha, bf16_t* __restrict__ o_mix) {
  extern __shared__ char smem_raw[];
  float*  s  = (float*)smem_raw;                                   // 16 x 2048 scores -> P
  bf16_t* qs = (bf16_t*)(smem_raw + 16 * 2048 * 4);                // 16 x 64 q tile
  float*  os = (float*)(smem_raw + 16 * 2048 * 4 + 16 * 64 * 2);   // 16 x 64 sparse out

  const int tid  = threadIdx.x;
  const int wave = tid >> 5, lane = tid & 31;
  const int half = lane >> 4, l16 = lane & 15;
  const int blk = blockIdx.x;
  const int bh = blk >> 7, tn = blk & 127;   // 128 row-tiles per head
  const int n0 = tn * 16;
  const int b = bh >> 4, h = bh & 15;
  const bf16_t* qhb = q_bf + (size_t)bh * NN * DD;
  const bf16_t* khb = k_bf + (size_t)bh * NN * DD;
  const bf16_t* vT  = vTb  + (size_t)bh * DD * NN;

  // stage q tile (16 rows x 64 cols, bf16) via the Tensor Data Mover
#if __has_builtin(__builtin_amdgcn_tensor_load_to_lds)
  if (wave == 0) {
    const unsigned lds_q = (unsigned)(uintptr_t)qs;     // wave-relative LDS addr
    const u64 ga = (u64)(uintptr_t)(qhb + (size_t)n0 * DD);
    // D# group0: count=1 | lds_addr | global_addr[56:0] | type=2
    u32x4 g0 = {1u, lds_q, (unsigned)(ga & 0xffffffffu),
                (unsigned)((ga >> 32) & 0x01ffffffu) | (2u << 30)};
    // D# group1: data_size=2B; tensor 64 x 2048, tile 64 x 16, stride0=64
    i32x8 g1 = {(int)(1u << 16),                         // data_size=1 (2 bytes)
                (int)((64u & 0xffffu) << 16),            // tensor_dim0 lo -> [31:16]
                (int)((2048u & 0xffffu) << 16),          // dim0 hi=0 | tensor_dim1 lo
                (int)(64u << 16),                        // dim1 hi=0 | tile_dim0=64
                (int)16,                                 // tile_dim1=16, tile_dim2=0
                (int)64,                                 // tensor_dim0_stride lo
                0, 0};                                   // stride0 hi, stride1 (unused)
    i32x4 gz4 = {0, 0, 0, 0};
    i32x8 gz8 = {0, 0, 0, 0, 0, 0, 0, 0};
    __builtin_amdgcn_tensor_load_to_lds(g0, g1, gz4, gz4, gz8, 0);
    __builtin_amdgcn_s_wait_tensorcnt((short)0);
  }
#else
  for (int idx = tid; idx < 16 * 64; idx += 128)
    qs[idx] = qhb[(size_t)(n0 + (idx >> 6)) * DD + (idx & 63)];
#endif
  __syncthreads();

  // phase 1: scores = (q @ k^T) / 8, via WMMA; 32 column-tiles per wave
  for (int ct = wave; ct < 128; ct += 4) {
    int m0 = ct * 16;
    v8f acc = {};
#pragma unroll
    for (int dk = 0; dk < DD; dk += 32) {
      v16bf af;
      const bf16_t* ap = qs + l16 * DD + dk;
#pragma unroll
      for (int j = 0; j < 16; ++j) af[j] = ap[16 * half + j];
      v16bf bfm;   // B[k=d][col=m] = k[m][d]: contiguous 16 bf16 per lane
      const bf16_t* bp = khb + (size_t)(m0 + l16) * DD + dk + 16 * half;
#pragma unroll
      for (int j = 0; j < 16; ++j) bfm[j] = bp[j];
      acc = __builtin_amdgcn_wmma_f32_16x16x32_bf16(
          false, af, false, bfm, (short)0, acc, false, false);
    }
#pragma unroll
    for (int r = 0; r < 8; ++r)
      s[(r + 8 * half) * 2048 + m0 + l16] = acc[r] * 0.125f;
  }
  __syncthreads();

  // phase 2: per-row softmax stats + exact 102nd-largest threshold (binary
  // search on order-preserving bits; softmax is monotone => threshold scores)
  for (int rr = 0; rr < 4; ++rr) {
    const int row = wave + 4 * rr;
    float* srow = s + row * 2048;
    float vmx = -3.4e38f, vmn = 3.4e38f;
    for (int j = lane; j < 2048; j += 32) {
      float x = srow[j];
      vmx = fmaxf(vmx, x); vmn = fminf(vmn, x);
    }
    vmx = wave_max_f(vmx); vmn = wave_min_f(vmn);
    unsigned lo = f2ord(vmn), hi = f2ord(vmx);
    while (lo < hi) {                       // largest t with count(s>=t) >= TOPK
      unsigned mid = lo + ((hi - lo + 1) >> 1);
      float t = ord2f(mid);
      int cnt = 0;
      for (int j = lane; j < 2048; j += 32) cnt += (srow[j] >= t) ? 1 : 0;
      cnt = wave_sum_i(cnt);
      if (cnt >= TOPK) lo = mid; else hi = mid - 1;
    }
    float kth = ord2f(lo);
    float esA = 0.f, esM = 0.f;
    for (int j = lane; j < 2048; j += 32) {
      float e = __expf(srow[j] - vmx);
      esA += e;
      if (srow[j] >= kth) esM += e;
    }
    esA = wave_sum_f(esA); esM = wave_sum_f(esM);
    // aws = aw*mask / (sum(aw*mask) + 1e-8)  with aw = e/Z  ==> e/(Em + 1e-8*Z)
    float denom = 1.0f / (esM + 1e-8f * esA);
    for (int j = lane; j < 2048; j += 32) {
      float x = srow[j];
      srow[j] = (x >= kth) ? __expf(x - vmx) * denom : 0.0f;   // P in place
    }
  }
  __syncthreads();

  // phase 3: o_s = P (16x2048) @ V (2048x64) via WMMA, one 16x16 tile / wave
  {
    const int e0 = 16 * wave;
    v8f acc = {};
    for (int mt = 0; mt < 64; ++mt) {
      int mk0 = mt * 32;
      v16bf af;   // convert masked probs f32 -> bf16 on fragment load
      const float* ap = s + l16 * 2048 + mk0;
#pragma unroll
      for (int j = 0; j < 16; ++j)
        af[j] = (bf16_t)ap[16 * half + j];
      v16bf bfm;  // B[k=m][col=e] = vT[e][m]: contiguous per lane
      const bf16_t* bp = vT + (size_t)(e0 + l16) * NN + mk0 + 16 * half;
#pragma unroll
      for (int j = 0; j < 16; ++j) bfm[j] = bp[j];
      acc = __builtin_amdgcn_wmma_f32_16x16x32_bf16(
          false, af, false, bfm, (short)0, acc, false, false);
    }
#pragma unroll
    for (int r = 0; r < 8; ++r)
      os[(r + 8 * half) * 64 + e0 + l16] = acc[r];
  }
  __syncthreads();

  // phase 4: linear branch o_l = qf@kv / (qf.kfsum + 1e-6), sigmoid mix
  const float a = 1.0f / (1.0f + __expf(-alpha[h]));
  const float* kvh  = kv + (size_t)bh * DD * DD;
  const float* ksum = kfsum + (size_t)bh * DD;
  for (int idx = tid; idx < 16 * 64; idx += 128) {
    int row = idx >> 6, e = idx & 63;
    int n = n0 + row;
    const float* qfr = qf + ((size_t)bh * NN + n) * DD;
    float z = 0.f, ol = 0.f;
#pragma unroll 8
    for (int d = 0; d < DD; ++d) {
      float qd = qfr[d];
      z  += qd * ksum[d];
      ol += qd * kvh[d * DD + e];
    }
    ol /= (z + 1e-6f);
    float o = a * os[idx] + (1.0f - a) * ol;
    o_mix[((size_t)b * NN + n) * CC + h * DD + e] = (bf16_t)o;
  }
}

// ---------------- K5: output projection GEMM (async + bf16 WMMA) ---------
__global__ __launch_bounds__(256)
void proj_gemm_kernel(const bf16_t* __restrict__ Xb, const bf16_t* __restrict__ Wb,
                      const float* __restrict__ bias, float* __restrict__ out) {
  __shared__ __align__(16) bf16_t As[2][128 * 32];
  __shared__ __align__(16) bf16_t Bs[2][128 * 32];
  const int tid  = threadIdx.x;
  const int wave = tid >> 5, lane = tid & 31;
  const int half = lane >> 4, l16 = lane & 15;
  const int rowBase = blockIdx.x * 128;
  const int colBase = blockIdx.y * 128;
  v8f acc[8] = {};
  const int cr = tid >> 1;
  const int ch = (tid & 1) * 16;
  const bf16_t* gA = Xb + (size_t)(rowBase + cr) * CC + ch;
  const bf16_t* gB = Wb + (size_t)(colBase + cr) * CC + ch;
  const unsigned ldsA0 = (unsigned)(uintptr_t)(&As[0][cr * 32 + ch]);
  const unsigned ldsB0 = (unsigned)(uintptr_t)(&Bs[0][cr * 32 + ch]);
  const unsigned BUFB = 128u * 32u * 2u;

  async_copy_b128(ldsA0,      (u64)(uintptr_t)gA);
  async_copy_b128(ldsA0 + 16, (u64)(uintptr_t)(gA + 8));
  async_copy_b128(ldsB0,      (u64)(uintptr_t)gB);
  async_copy_b128(ldsB0 + 16, (u64)(uintptr_t)(gB + 8));

  for (int it = 0; it < 32; ++it) {
    const int buf = it & 1;
    if (it + 1 < 32) {
      const bf16_t* nA = gA + (size_t)(it + 1) * 32;
      const bf16_t* nB = gB + (size_t)(it + 1) * 32;
      const unsigned la = ldsA0 + (unsigned)(buf ^ 1) * BUFB;
      const unsigned lb = ldsB0 + (unsigned)(buf ^ 1) * BUFB;
      async_copy_b128(la,      (u64)(uintptr_t)nA);
      async_copy_b128(la + 16, (u64)(uintptr_t)(nA + 8));
      async_copy_b128(lb,      (u64)(uintptr_t)nB);
      async_copy_b128(lb + 16, (u64)(uintptr_t)(nB + 8));
      wait_async_le4();
    } else {
      wait_async_le0();
    }
    __syncthreads();
    v16bf af;
    const bf16_t* ap = &As[buf][(16 * wave + l16) * 32];
#pragma unroll
    for (int j = 0; j < 16; ++j) af[j] = ap[16 * half + j];
#pragma unroll
    for (int ct = 0; ct < 8; ++ct) {
      v16bf bfm;
      const bf16_t* bp = &Bs[buf][(16 * ct + l16) * 32 + 16 * half];
#pragma unroll
      for (int j = 0; j < 16; ++j) bfm[j] = bp[j];
      acc[ct] = __builtin_amdgcn_wmma_f32_16x16x32_bf16(
          false, af, false, bfm, (short)0, acc[ct], false, false);
    }
    __syncthreads();
  }
#pragma unroll
  for (int ct = 0; ct < 8; ++ct) {
#pragma unroll
    for (int r = 0; r < 8; ++r) {
      int row_g = rowBase + 16 * wave + r + 8 * half;
      int col_g = colBase + 16 * ct + l16;
      out[(size_t)row_g * CC + col_g] = acc[ct][r] + bias[col_g];
    }
  }
}

// ---------------- host orchestration ----------------
extern "C" void kernel_launch(void* const* d_in, const int* in_sizes, int n_in,
                              void* d_out, int out_size, void* d_ws, size_t ws_size,
                              hipStream_t stream) {
  (void)in_sizes; (void)n_in; (void)out_size; (void)ws_size;
  const float* x     = (const float*)d_in[0];
  const float* Wqkv  = (const float*)d_in[1];
  const float* bqkv  = (const float*)d_in[2];
  const float* qg    = (const float*)d_in[3];
  const float* qb    = (const float*)d_in[4];
  const float* kg    = (const float*)d_in[5];
  const float* kb    = (const float*)d_in[6];
  const float* alpha = (const float*)d_in[7];
  const float* Wproj = (const float*)d_in[8];
  const float* bproj = (const float*)d_in[9];
  float* out = (float*)d_out;

  // workspace carving (all sizes 256B multiples)
  char* ws = (char*)d_ws;
  size_t off = 0;
  auto carve = [&](size_t bytes) -> char* {
    char* p = ws + off;
    off += (bytes + 255) & ~(size_t)255;
    return p;
  };
  const size_t E = (size_t)BB * HH * NN * DD;      // 4,194,304 elements
  bf16_t* xb     = (bf16_t*)carve((size_t)BB * NN * CC * 2);
  bf16_t* wqkvb  = (bf16_t*)carve((size_t)3 * CC * CC * 2);
  bf16_t* wprojb = (bf16_t*)carve((size_t)CC * CC * 2);
  float*  qbuf   = (float*)carve(E * 4);
  float*  kbuf   = (float*)carve(E * 4);
  float*  vbuf   = (float*)carve(E * 4);
  bf16_t* qbf    = (bf16_t*)carve(E * 2);
  bf16_t* kbf    = (bf16_t*)carve(E * 2);
  bf16_t* vTb    = (bf16_t*)carve(E * 2);
  float*  qf     = (float*)carve(E * 4);
  float*  kf     = (float*)carve(E * 4);
  float*  kfsum  = (float*)carve((size_t)BB * HH * DD * 4);
  float*  kvb    = (float*)carve((size_t)BB * HH * DD * DD * 4);
  bf16_t* omix   = (bf16_t*)carve((size_t)BB * NN * CC * 2);

  zero_f32_kernel<<<8, 256, 0, stream>>>(kfsum, BB * HH * DD);
  cvt_f32_to_bf16_kernel<<<2048, 256, 0, stream>>>(x, xb, BB * NN * CC);
  cvt_f32_to_bf16_kernel<<<2048, 256, 0, stream>>>(Wqkv, wqkvb, 3 * CC * CC);
  cvt_f32_to_bf16_kernel<<<1024, 256, 0, stream>>>(Wproj, wprojb, CC * CC);

  qkv_gemm_kernel<<<dim3(32, 24), 256, 0, stream>>>(xb, wqkvb, bqkv,
                                                    qbuf, kbuf, vbuf, vTb);

  ln_feat_kernel<<<(BB * HH * NN) / 8, 256, 0, stream>>>(qbuf, qg, qb, qbf, qf, nullptr, 0);
  ln_feat_kernel<<<(BB * HH * NN) / 8, 256, 0, stream>>>(kbuf, kg, kb, kbf, kf, kfsum, 1);

  kv_kernel<<<BB * HH, 256, 0, stream>>>(kf, vbuf, kvb);

  (void)hipFuncSetAttribute((const void*)sparse_attn_kernel,
                            hipFuncAttributeMaxDynamicSharedMemorySize, SPARSE_SMEM);
  sparse_attn_kernel<<<BB * HH * (NN / 16), 128, SPARSE_SMEM, stream>>>(
      qbf, kbf, vTb, qf, kvb, kfsum, alpha, omix);

  proj_gemm_kernel<<<dim3(32, 8), 256, 0, stream>>>(omix, wprojb, bproj, out);
}
